// SparsemaxAttention_26302379720967
// MI455X (gfx1250) — compile-verified, matrix operand order checked by reading, not run on hardware
//
#include <hip/hip_runtime.h>
#include <hip/hip_bf16.h>

// SparsemaxAttention for MI455X (gfx1250), wave32 + WMMA bf16.
//
// Pipeline:
//   1) gemm_wmma<XF32=1,OUTF32=0,REMAP=0> x3 : Q/K/V = x@W.T+b  (f32 in, bf16 out,
//      rows permuted (a*8+b) -> b*4096+a so each (b,a) position is contiguous)
//   2) attn_sparsemax : per (b,a) 16x16 head-mixing scores, sparsemax(alpha=1.5),
//      attn@V ; writes attn_out bf16 in (B,H,A,hd) order == (B*4096,1024) row-major
//   3) gemm_wmma<XF32=0,OUTF32=1,REMAP=1> : out = AO@Wo.T+bo, store row
//      (b*4096+r) -> r*8+b  == final swapaxes(0,1)
//
// Workspace: Q,K,V,AO bf16 = 4 x 64MB = 256MB.

#define SEQ   4096
#define BATCH 8
#define DM    1024
#define MROWS (SEQ*BATCH)   // 32768

typedef __attribute__((ext_vector_type(16))) __bf16 v16bf;
typedef __attribute__((ext_vector_type(8)))  float  v8f;

union FragBF {
    v16bf v;
    uint4 u[2];
    unsigned short s[16];
};

// native f32 -> bf16 (round-to-nearest-even via hardware cvt)
__device__ __forceinline__ unsigned short f2bf(float f) {
    union { __bf16 h; unsigned short s; } c;
    c.h = (__bf16)f;
    return c.s;
}
__device__ __forceinline__ float bf_lo(unsigned int u) {
    union { unsigned int u; float f; } c; c.u = u << 16; return c.f;
}
__device__ __forceinline__ float bf_hi(unsigned int u) {
    union { unsigned int u; float f; } c; c.u = u & 0xffff0000u; return c.f;
}

// ---------------------------------------------------------------------------
// C[M=32768, N=1024] = X(MxK=1024) * W^T + bias.
// Block = 256 threads (8 waves), tile 128(M) x 64(N), K-chunk 32.
// Wave w owns M-strip [w*16, w*16+16), computes 4 accumulators (N tiles of 16).
// W tile (64x32) double-buffered in LDS as bf16 (8x intra-block reuse, global
// weight loads overlapped with WMMA); A fragments loaded straight from global
// (no intra-block reuse; X/AO stay hot in the 192MB L2).
// ---------------------------------------------------------------------------
template<bool XF32, bool OUTF32, int REMAP>
__global__ __launch_bounds__(256) void gemm_wmma(
    const void* __restrict__ Xv, const float* __restrict__ W,
    const float* __restrict__ bias, void* __restrict__ Outv)
{
    // 64 rows x 32 cols bf16 per buffer, row stride 40 elems (80B) -> the 16
    // rows of a b128 fragment load touch all 64 LDS banks exactly once.
    __shared__ __align__(16) unsigned short Blds[2][64 * 40];

    const int tid   = threadIdx.x;
    const int lane  = tid & 31;
    const int wv    = tid >> 5;
    const int l15   = lane & 15;
    const int hi    = lane >> 4;            // 0: lanes 0-15, 1: lanes 16-31
    const int Mbase = blockIdx.x * 128;
    const int Nbase = blockIdx.y * 64;
    const int aRow  = Mbase + wv * 16 + l15;

    v8f acc[4];
    const v8f vzero = {0.f, 0.f, 0.f, 0.f, 0.f, 0.f, 0.f, 0.f};
#pragma unroll
    for (int t = 0; t < 4; ++t) acc[t] = vzero;

    // B staging assignment: thread -> (row r, 8 cols starting at c0)
    const int r  = tid >> 2;
    const int c0 = (tid & 3) * 8;
    const float* wrow = W + (size_t)(Nbase + r) * DM + c0;

    auto stageB = [&](int kc, int buf) {
        float4 w0 = *(const float4*)(wrow + kc);
        float4 w1 = *(const float4*)(wrow + kc + 4);
        union { unsigned short s[8]; uint4 q; } pk;
        pk.s[0] = f2bf(w0.x); pk.s[1] = f2bf(w0.y);
        pk.s[2] = f2bf(w0.z); pk.s[3] = f2bf(w0.w);
        pk.s[4] = f2bf(w1.x); pk.s[5] = f2bf(w1.y);
        pk.s[6] = f2bf(w1.z); pk.s[7] = f2bf(w1.w);
        *(uint4*)&Blds[buf][r * 40 + c0] = pk.q;
    };

    stageB(0, 0);

    for (int kc = 0; kc < DM; kc += 32) {
        const int cur = (kc >> 5) & 1;
        __syncthreads();                     // staging of `cur` visible
        if (kc + 32 < DM) stageB(kc + 32, cur ^ 1);   // overlap next chunk

        // A fragment: 16x32 bf16, ISA layout: lanes 0-15 K{0..7,16..23},
        // lanes 16-31 K{8..15,24..31}, M = lane%16 within the wave strip.
        FragBF a;
        if constexpr (XF32) {
            const float* ap = (const float*)Xv + (size_t)aRow * DM + kc + hi * 8;
            float4 p0 = *(const float4*)(ap);
            float4 p1 = *(const float4*)(ap + 4);
            float4 p2 = *(const float4*)(ap + 16);
            float4 p3 = *(const float4*)(ap + 20);
            a.s[0]  = f2bf(p0.x); a.s[1]  = f2bf(p0.y);
            a.s[2]  = f2bf(p0.z); a.s[3]  = f2bf(p0.w);
            a.s[4]  = f2bf(p1.x); a.s[5]  = f2bf(p1.y);
            a.s[6]  = f2bf(p1.z); a.s[7]  = f2bf(p1.w);
            a.s[8]  = f2bf(p2.x); a.s[9]  = f2bf(p2.y);
            a.s[10] = f2bf(p2.z); a.s[11] = f2bf(p2.w);
            a.s[12] = f2bf(p3.x); a.s[13] = f2bf(p3.y);
            a.s[14] = f2bf(p3.z); a.s[15] = f2bf(p3.w);
        } else {
            const unsigned short* ap =
                (const unsigned short*)Xv + (size_t)aRow * DM + kc + hi * 8;
            a.u[0] = *(const uint4*)(ap);
            a.u[1] = *(const uint4*)(ap + 16);
        }

        // Load ALL four B fragments first so the ds_loads pipeline, then
        // issue the four WMMAs back-to-back.
        FragBF b[4];
#pragma unroll
        for (int t = 0; t < 4; ++t) {
            const unsigned short* bp = &Blds[cur][(t * 16 + l15) * 40 + hi * 8];
            b[t].u[0] = *(const uint4*)(bp);
            b[t].u[1] = *(const uint4*)(bp + 16);
        }
#pragma unroll
        for (int t = 0; t < 4; ++t) {
            acc[t] = __builtin_amdgcn_wmma_f32_16x16x32_bf16(
                false, a.v, false, b[t].v, (short)0, acc[t], false, false);
        }
    }

    // Epilogue: C layout VGPR v, lanes 0-15: (M=v, N=lane); 16-31: (M=v+8).
#pragma unroll
    for (int t = 0; t < 4; ++t) {
        const int   col = Nbase + t * 16 + l15;
        const float bv  = bias[col];
#pragma unroll
        for (int v = 0; v < 8; ++v) {
            const int m   = Mbase + wv * 16 + v + hi * 8;
            const float val = acc[t][v] + bv;
            size_t row;
            if constexpr (REMAP == 0)       // (a*8+b) -> b*4096+a
                row = (size_t)(m & 7) * SEQ + (m >> 3);
            else                            // (b*4096+r) -> r*8+b
                row = (size_t)(m & (SEQ - 1)) * BATCH + (m >> 12);
            if constexpr (OUTF32)
                ((float*)Outv)[row * DM + col] = val;
            else
                ((unsigned short*)Outv)[row * DM + col] = f2bf(val);
        }
    }
}

// ---------------------------------------------------------------------------
// Per-position head-mixing sparsemax attention. Block = 256 threads handles
// 16 positions; thread (pos, h) owns one 16-wide score row. K/V rows for the
// block's positions staged in LDS (2 x 32KB bf16).
// Sort-free sparsemax: i in support  <=>  z_i * k_i > S_i - 1, where
// S_i/k_i = sum/count of elements >= z_i (max element always passes => k>=1).
// ---------------------------------------------------------------------------
__global__ __launch_bounds__(256) void attn_sparsemax(
    const unsigned short* __restrict__ Qws, const unsigned short* __restrict__ Kws,
    const unsigned short* __restrict__ Vws, unsigned short* __restrict__ AOws)
{
    __shared__ __align__(16) unsigned short Ksh[16 * DM];
    __shared__ __align__(16) unsigned short Vsh[16 * DM];

    const int tid = threadIdx.x;
    const int p0  = blockIdx.x * 16;

#pragma unroll
    for (int j = 0; j < 8; ++j) {          // 2048 uint4 chunks per matrix
        const int c   = tid + j * 256;
        const int pos = c >> 7;
        const int off = (c & 127) * 8;
        *(uint4*)&Ksh[pos * DM + off] =
            *(const uint4*)&Kws[(size_t)(p0 + pos) * DM + off];
        *(uint4*)&Vsh[pos * DM + off] =
            *(const uint4*)&Vws[(size_t)(p0 + pos) * DM + off];
    }
    __syncthreads();

    const int pos = tid >> 4;
    const int h   = tid & 15;
    const int p   = p0 + pos;

    // Q row (64 bf16 = 32 packed dwords) kept in registers.
    unsigned int q[32];
    {
        const uint4* qp = (const uint4*)&Qws[(size_t)p * DM + h * 64];
#pragma unroll
        for (int i = 0; i < 8; ++i) ((uint4*)q)[i] = qp[i];
    }

    // scores[g] = (Q_h . K_g) / sqrt(64) / alpha
    const float scl = 0.125f / 1.5f;
    float z[16];
#pragma unroll
    for (int g = 0; g < 16; ++g) {
        const unsigned int* kp = (const unsigned int*)&Ksh[pos * DM + g * 64];
        float acc = 0.f;
#pragma unroll
        for (int i = 0; i < 32; ++i) {
            const unsigned int qu = q[i], ku = kp[i];
            acc += bf_lo(qu) * bf_lo(ku);
            acc += bf_hi(qu) * bf_hi(ku);
        }
        z[g] = acc * scl;
    }

    // sparsemax (pivot form, no sort)
    float tsum = 0.f, kcnt = 0.f;
#pragma unroll
    for (int i = 0; i < 16; ++i) {
        float Si = 0.f, ki = 0.f;
#pragma unroll
        for (int j = 0; j < 16; ++j) {
            const bool ge = z[j] >= z[i];
            Si += ge ? z[j] : 0.f;
            ki += ge ? 1.f : 0.f;
        }
        const bool sup = z[i] * ki > (Si - 1.f);
        tsum += sup ? z[i] : 0.f;
        kcnt += sup ? 1.f : 0.f;
    }
    const float tau = (tsum - 1.f) / kcnt;
    float attn[16];
#pragma unroll
    for (int i = 0; i < 16; ++i) attn[i] = fmaxf(z[i] - tau, 0.f);

    // attn_out[h, d] = sum_g attn[g] * V[g, d], stored (B,H,A,hd) bf16
    const int b = p >> 12;
    const int a = p & (SEQ - 1);
    unsigned short* op = &AOws[(((size_t)b * 16 + h) * SEQ + a) * 64];
#pragma unroll
    for (int dc = 0; dc < 8; ++dc) {       // 8 columns at a time
        float o[8] = {0.f, 0.f, 0.f, 0.f, 0.f, 0.f, 0.f, 0.f};
#pragma unroll
        for (int g = 0; g < 16; ++g) {
            const unsigned int* vp =
                (const unsigned int*)&Vsh[pos * DM + g * 64 + dc * 8];
            const float ag = attn[g];
#pragma unroll
            for (int i = 0; i < 4; ++i) {
                const unsigned int vu = vp[i];
                o[2 * i]     += ag * bf_lo(vu);
                o[2 * i + 1] += ag * bf_hi(vu);
            }
        }
        union { unsigned short s[8]; uint4 qv; } pk;
#pragma unroll
        for (int i = 0; i < 8; ++i) pk.s[i] = f2bf(o[i]);
        *(uint4*)&op[dc * 8] = pk.qv;
    }
}

// ---------------------------------------------------------------------------
extern "C" void kernel_launch(void* const* d_in, const int* in_sizes, int n_in,
                              void* d_out, int out_size, void* d_ws, size_t ws_size,
                              hipStream_t stream) {
    const float* query = (const float*)d_in[0];
    const float* key_  = (const float*)d_in[1];
    const float* value = (const float*)d_in[2];
    const float* Wq = (const float*)d_in[3];  const float* bq = (const float*)d_in[4];
    const float* Wk = (const float*)d_in[5];  const float* bk = (const float*)d_in[6];
    const float* Wv = (const float*)d_in[7];  const float* bv = (const float*)d_in[8];
    const float* Wo = (const float*)d_in[9];  const float* bo = (const float*)d_in[10];
    float* out = (float*)d_out;

    const size_t MAT = (size_t)MROWS * DM;          // 32768*1024 elements
    unsigned short* Qws  = (unsigned short*)d_ws;   // bf16, 64MB each
    unsigned short* Kws  = Qws + MAT;
    unsigned short* Vws  = Kws + MAT;
    unsigned short* AOws = Vws + MAT;

    dim3 blk(256, 1, 1);
    dim3 gG(MROWS / 128, DM / 64, 1);               // (256, 16)

    gemm_wmma<true,  false, 0><<<gG, blk, 0, stream>>>(query, Wq, bq, Qws);
    gemm_wmma<true,  false, 0><<<gG, blk, 0, stream>>>(key_,  Wk, bk, Kws);
    gemm_wmma<true,  false, 0><<<gG, blk, 0, stream>>>(value, Wv, bv, Vws);

    attn_sparsemax<<<MROWS / 16, blk, 0, stream>>>(Qws, Kws, Vws, AOws);

    gemm_wmma<false, true,  1><<<gG, blk, 0, stream>>>(AOws, Wo, bo, out);
}
